// MemoryModel_35270271435207
// MI455X (gfx1250) — matrix operand order repaired and checked by reading, not compiled, not run on hardware
//
#include <hip/hip_runtime.h>

// ---------------------------------------------------------------------------
// MemoryModel on gfx1250 (MI455X): token-table + chunked delta-rule scan
// using V_WMMA_F32_16X16X32_BF16 (K-dim 32 == hidden dim H).
// ---------------------------------------------------------------------------

typedef __attribute__((ext_vector_type(16))) __bf16 v16bf;
typedef __attribute__((ext_vector_type(8)))  float  v8f;

constexpr int HD   = 32;     // hidden dim
constexpr int SEQL = 4096;   // sequence length
constexpr int NV   = 64;     // vocab
constexpr int NCH  = 256;    // chunks of 16 steps (last step = zero-key pad)

// LDS row strides (in elements) chosen for bank-conflict avoidance / alignment
constexpr int KSTR = 40;     // bf16 K-chunk rows (80B, 16B-aligned)
constexpr int MSTR = 36;     // f32 M rows (144B, 16B-aligned)
constexpr int VSTR = 21;     // f32 V rows
constexpr int GSTR = 17;     // f32 G rows
constexpr int DSTR = 33;     // f32 Delta rows

// workspace layout (bytes): hidf f32[65*32] @0 ; hidb bf16[65*32] @8320 ; ctx f32[256*32] @12480
constexpr int WS_HIDB = 8320;
constexpr int WS_CTX  = 12480;

// ------------------------- kernel 1: token table ---------------------------
// hid_table[v] = layernorm(embed[v] + relu(embed[v]@W1+b1)@W2+b2)  ; row 64 = 0
__global__ __launch_bounds__(64) void mm_build_table(
    const float* __restrict__ embed, const float* __restrict__ W1, const float* __restrict__ b1,
    const float* __restrict__ W2,    const float* __restrict__ b2,
    const float* __restrict__ gamma, const float* __restrict__ beta,
    float* __restrict__ hidf, unsigned short* __restrict__ hidb_u)
{
  __bf16* hidb = (__bf16*)hidb_u;
  const int v = threadIdx.x;          // one token per thread
  float h[HD], z2[HD];
#pragma unroll
  for (int i = 0; i < HD; i++) { h[i] = embed[v*HD + i]; z2[i] = b2[i]; }
  for (int j = 0; j < 2*HD; j++) {
    float a = b1[j];
#pragma unroll
    for (int i = 0; i < HD; i++) a += h[i] * W1[i*2*HD + j];
    a = fmaxf(a, 0.f);
#pragma unroll
    for (int i = 0; i < HD; i++) z2[i] += a * W2[j*HD + i];
  }
  float mu = 0.f;
#pragma unroll
  for (int i = 0; i < HD; i++) { z2[i] += h[i]; mu += z2[i]; }
  mu *= (1.f/HD);
  float var = 0.f;
#pragma unroll
  for (int i = 0; i < HD; i++) { float d = z2[i]-mu; var += d*d; }
  var *= (1.f/HD);
  const float rs = rsqrtf(var + 1e-5f);
#pragma unroll
  for (int i = 0; i < HD; i++) {
    float y = (z2[i]-mu)*rs*gamma[i] + beta[i];
    hidf[v*HD + i] = y;
    hidb[v*HD + i] = (__bf16)y;
  }
  if (v < HD) { hidf[NV*HD + v] = 0.f; hidb[NV*HD + v] = (__bf16)0.f; }  // zero-key row
}

// ---------------------- kernel 2: chunked delta scan -----------------------
// One wave per batch. M (32x32 f32) lives in 4 WMMA accumulator tiles.
// Per 16-step chunk: V = M K^T (2 wmma), G = K K^T (1 wmma), 16-step
// triangular solve for Delta, M += Delta^T K (4 wmma, K padded to 32).
__global__ __launch_bounds__(32) void mm_delta_scan(
    const int* __restrict__ seq, const float* __restrict__ hidf_g,
    const unsigned short* __restrict__ hidb_g, float* __restrict__ ctx)
{
  __shared__ float  HLF[(NV+1)*HD];          // f32 token table
  __shared__ __bf16 HLB[(NV+1)*HD];          // bf16 token table
  __shared__ __bf16 Klds[16*KSTR];           // chunk keys [t][j]
  __shared__ float  Mlds[32*MSTR];           // M spill [i][j]
  __shared__ float  Vlds[32*VSTR];           // V spill [i][t]
  __shared__ float  Glds[16*GSTR];           // Gram   [s][t]
  __shared__ float  Dlds[16*DSTR];           // Delta  [t][i]
  __shared__ int    TOK[16];

  const int lane = threadIdx.x;
  const int lh = lane & 15, hi = lane >> 4;
  const int b = blockIdx.x;

  for (int idx = lane; idx < (NV+1)*HD; idx += 32) {
    HLF[idx] = hidf_g[idx];
    HLB[idx] = ((const __bf16*)hidb_g)[idx];
  }
  __syncthreads();

  v8f macc[2][2];
#pragma unroll
  for (int ti = 0; ti < 2; ti++)
#pragma unroll
    for (int tj = 0; tj < 2; tj++)
#pragma unroll
      for (int r = 0; r < 8; r++) macc[ti][tj][r] = 0.f;

  const v8f zc = {0.f,0.f,0.f,0.f,0.f,0.f,0.f,0.f};
  const int* sb = seq + b*SEQL;

  for (int c = 0; c < NCH; c++) {
    // ---- tokens + K chunk (bf16 rows gathered from table) ----
    const int pos = (c << 4) + lh;
    const int tok = (pos < SEQL-1) ? sb[pos] : NV;   // pad -> zero row
    if (hi == 0) TOK[lh] = tok;
    {
      const unsigned int* src = (const unsigned int*)&HLB[tok*HD + hi*16];
      unsigned int*       dst = (unsigned int*)&Klds[lh*KSTR + hi*16];
#pragma unroll
      for (int w = 0; w < 8; w++) dst[w] = src[w];
    }
    if (c + 1 < NCH) __builtin_prefetch(sb + pos + 16);
    __syncthreads();

    // ---- spill M accumulators (C layout: lane holds (r+8*hi, lh)) ----
#pragma unroll
    for (int ti = 0; ti < 2; ti++)
#pragma unroll
      for (int tj = 0; tj < 2; tj++)
#pragma unroll
        for (int r = 0; r < 8; r++)
          Mlds[(ti*16 + r + 8*hi)*MSTR + tj*16 + lh] = macc[ti][tj][r];
    __syncthreads();

    // ---- fragments: B = K^T (K-dim j, N = t), A_K = K, A_M = M tiles ----
    v16bf bKT;
#pragma unroll
    for (int e = 0; e < 16; e++) bKT[e] = Klds[lh*KSTR + (e & 7) + 8*hi + 16*(e >> 3)];
    v16bf aK;
#pragma unroll
    for (int e = 0; e < 8; e++) {
      aK[e]   = Klds[lh*KSTR + hi*8 + e];
      aK[8+e] = Klds[lh*KSTR + 16 + hi*8 + e];
    }
    v16bf aM0, aM1;
#pragma unroll
    for (int e = 0; e < 8; e++) {
      aM0[e]   = (__bf16)Mlds[lh*MSTR + hi*8 + e];
      aM0[8+e] = (__bf16)Mlds[lh*MSTR + 16 + hi*8 + e];
      aM1[e]   = (__bf16)Mlds[(16+lh)*MSTR + hi*8 + e];
      aM1[8+e] = (__bf16)Mlds[(16+lh)*MSTR + 16 + hi*8 + e];
    }
    // V = M K^T ; G = K K^T
    v8f vac0 = __builtin_amdgcn_wmma_f32_16x16x32_bf16(false, aM0, false, bKT, (short)0, zc, false, false);
    v8f vac1 = __builtin_amdgcn_wmma_f32_16x16x32_bf16(false, aM1, false, bKT, (short)0, zc, false, false);
    v8f gac  = __builtin_amdgcn_wmma_f32_16x16x32_bf16(false, aK,  false, bKT, (short)0, zc, false, false);
#pragma unroll
    for (int r = 0; r < 8; r++) {
      Vlds[(r + 8*hi)*VSTR + lh]      = vac0[r];
      Vlds[(16 + r + 8*hi)*VSTR + lh] = vac1[r];
      Glds[(r + 8*hi)*GSTR + lh]      = gac[r];
    }
    __syncthreads();

    // ---- triangular solve: lane = hidden index i, fully unrolled in t ----
    float acc[16];
#pragma unroll
    for (int t = 0; t < 16; t++) acc[t] = Vlds[lane*VSTR + t];
#pragma unroll
    for (int t = 0; t < 16; t++) {
      const int   tk  = TOK[t];                      // uniform broadcast read
      const float kti = HLF[tk*HD + lane];
      const float d   = Glds[t*GSTR + t] + 1e-6f;
      const float delta = kti - acc[t] * __builtin_amdgcn_rcpf(d);
      Dlds[t*DSTR + lane] = delta;
#pragma unroll
      for (int u = t + 1; u < 16; u++) acc[u] += delta * Glds[t*GSTR + u];
    }
    __syncthreads();

    // ---- M += Delta^T K (K-dim 16, zero-padded to 32 in A) ----
    v16bf aD[2];
#pragma unroll
    for (int ti = 0; ti < 2; ti++)
#pragma unroll
      for (int e = 0; e < 8; e++) {
        aD[ti][e]   = (__bf16)Dlds[(e + 8*hi)*DSTR + ti*16 + lh];
        aD[ti][8+e] = (__bf16)0.f;                   // K = 16..31 -> 0
      }
    v16bf bK2[2];
#pragma unroll
    for (int tj = 0; tj < 2; tj++)
#pragma unroll
      for (int e = 0; e < 16; e++)                   // k = e + 8*hi for e < 8; e >= 8 don't-care (A==0 there)
        bK2[tj][e] = Klds[((e & 7) + 8*hi)*KSTR + tj*16 + lh];
#pragma unroll
    for (int ti = 0; ti < 2; ti++)
#pragma unroll
      for (int tj = 0; tj < 2; tj++)
        macc[ti][tj] = __builtin_amdgcn_wmma_f32_16x16x32_bf16(
            false, aD[ti], false, bK2[tj], (short)0, macc[ti][tj], false, false);
    __syncthreads();
  }

  // ---- context = M @ q, q = table[seq[b, L-1]] ----
#pragma unroll
  for (int ti = 0; ti < 2; ti++)
#pragma unroll
    for (int tj = 0; tj < 2; tj++)
#pragma unroll
      for (int r = 0; r < 8; r++)
        Mlds[(ti*16 + r + 8*hi)*MSTR + tj*16 + lh] = macc[ti][tj][r];
  __syncthreads();
  const int tq = sb[SEQL-1];
  float cacc = 0.f;
#pragma unroll 8
  for (int j = 0; j < HD; j++) cacc += Mlds[lane*MSTR + j] * HLF[tq*HD + j];
  ctx[b*HD + lane] = cacc;
}

// -------------------- kernel 3: (ctx@Wr+br)@Wo+bo via WMMA -----------------
__global__ __launch_bounds__(32) void mm_out_proj(
    const float* __restrict__ ctx, const float* __restrict__ Wr, const float* __restrict__ br,
    const float* __restrict__ Wo,  const float* __restrict__ bo, float* __restrict__ out)
{
  __shared__ float T1[16*MSTR];
  const int lane = threadIdx.x, lh = lane & 15, hi = lane >> 4, rb = blockIdx.x;
  const v8f zc = {0.f,0.f,0.f,0.f,0.f,0.f,0.f,0.f};

  // A = 16 rows of context (f32 -> bf16)
  const float* crow = ctx + (rb*16 + lh)*HD;
  v16bf aC;
#pragma unroll
  for (int e = 0; e < 8; e++) {
    aC[e]   = (__bf16)crow[hi*8 + e];
    aC[8+e] = (__bf16)crow[16 + hi*8 + e];
  }
  // t1 = ctx @ Wr + br   (2 col tiles, K = 32)
#pragma unroll
  for (int tj = 0; tj < 2; tj++) {
    v16bf bW;
#pragma unroll
    for (int e = 0; e < 16; e++) bW[e] = (__bf16)Wr[((e & 7) + 8*hi + 16*(e >> 3))*HD + tj*16 + lh];
    v8f t1 = __builtin_amdgcn_wmma_f32_16x16x32_bf16(false, aC, false, bW, (short)0, zc, false, false);
    const float bias = br[tj*16 + lh];
#pragma unroll
    for (int r = 0; r < 8; r++) T1[(r + 8*hi)*MSTR + tj*16 + lh] = t1[r] + bias;
  }
  __syncthreads();
  v16bf aT;
#pragma unroll
  for (int e = 0; e < 8; e++) {
    aT[e]   = (__bf16)T1[lh*MSTR + hi*8 + e];
    aT[8+e] = (__bf16)T1[lh*MSTR + 16 + hi*8 + e];
  }
  // out = t1 @ Wo + bo   (4 col tiles, K = 32)
#pragma unroll
  for (int tj = 0; tj < 4; tj++) {
    v16bf bW;
#pragma unroll
    for (int e = 0; e < 16; e++) bW[e] = (__bf16)Wo[((e & 7) + 8*hi + 16*(e >> 3))*NV + tj*16 + lh];
    v8f o = __builtin_amdgcn_wmma_f32_16x16x32_bf16(false, aT, false, bW, (short)0, zc, false, false);
    const float bias = bo[tj*16 + lh];
#pragma unroll
    for (int r = 0; r < 8; r++)
      out[(rb*16 + r + 8*hi)*NV + tj*16 + lh] = o[r] + bias;
  }
}

// ------------------------------- launcher ----------------------------------
extern "C" void kernel_launch(void* const* d_in, const int* in_sizes, int n_in,
                              void* d_out, int out_size, void* d_ws, size_t ws_size,
                              hipStream_t stream)
{
  const int*   seq   = (const int*)  d_in[0];
  const float* embed = (const float*)d_in[1];
  const float* W1    = (const float*)d_in[2];
  const float* b1    = (const float*)d_in[3];
  const float* W2    = (const float*)d_in[4];
  const float* b2    = (const float*)d_in[5];
  const float* gam   = (const float*)d_in[6];
  const float* bet   = (const float*)d_in[7];
  const float* Wr    = (const float*)d_in[8];
  const float* br    = (const float*)d_in[9];
  const float* Wo    = (const float*)d_in[10];
  const float* bo    = (const float*)d_in[11];
  float* out = (float*)d_out;

  char* ws = (char*)d_ws;
  float*          hidf = (float*)ws;
  unsigned short* hidb = (unsigned short*)(ws + WS_HIDB);
  float*          ctx  = (float*)(ws + WS_CTX);

  mm_build_table<<<1, 64, 0, stream>>>(embed, W1, b1, W2, b2, gam, bet, hidf, hidb);
  mm_delta_scan<<<256, 32, 0, stream>>>(seq, hidf, hidb, ctx);
  mm_out_proj<<<16, 32, 0, stream>>>(ctx, Wr, br, Wo, bo, out);
}
